// QuantizedBottleneck_60017872995060
// MI455X (gfx1250) — 8-bit matrix operand order repaired and checked by reading, compiled, not run on hardware
//
#include <hip/hip_runtime.h>

typedef __attribute__((ext_vector_type(8))) int v8i;

#define OFF_W1 0
#define OFF_W2 16384
#define OFF_W3 53248
#define OFF_BN 69632
#define OFF_Y1 73728
#define OFF_Y2 6496256
#define OFF_RA 12918784
#define BN1W 0
#define BN1B 64
#define BN2W 128
#define BN2B 192
#define BN3W 256
#define BN3B 512

__device__ __forceinline__ int m0shift32(int acc, int M0, int shift) {
  unsigned lo = (unsigned)acc * (unsigned)M0;
  int ph = __mulhi(acc, M0);
  unsigned nudge = (acc >= 0) ? 0x40000000u : 0xC0000001u;
  unsigned lo2 = lo + nudge;
  int hi2 = ph + (acc >> 31) + (lo2 < lo ? 1 : 0);
#if __has_builtin(__builtin_amdgcn_alignbit)
  int hi = (int)__builtin_amdgcn_alignbit((unsigned)hi2, lo2, 31u);
#else
  int hi = (int)(((unsigned)hi2 << 1) | (lo2 >> 31));
#endif
  return (hi + (1 << (shift - 1))) >> shift;
}
__device__ __forceinline__ int rq32(int acc, int M0, int shift, int zout) {
  int v = m0shift32(acc, M0, shift) + zout;
  v = v < 0 ? 0 : v;
  return v > 255 ? 255 : v;
}

__device__ __forceinline__ v8i load_a_frag(const signed char* row, int h) {
  int2 a01 = *(const int2*)(row + 8 * h);
  int2 a23 = *(const int2*)(row + 16 + 8 * h);
  int2 a45 = *(const int2*)(row + 32 + 8 * h);
  int2 a67 = *(const int2*)(row + 48 + 8 * h);
  v8i a;
  a[0] = a01.x; a[1] = a01.y; a[2] = a23.x; a[3] = a23.y;
  a[4] = a45.x; a[5] = a45.y; a[6] = a67.x; a[7] = a67.y;
  return a;
}
__device__ __forceinline__ v8i load_b_frag(const signed char* col, int h) {
  int2 b01 = *(const int2*)(col + 8 * h);
  int2 b23 = *(const int2*)(col + 16 + 8 * h);
  int2 b45 = *(const int2*)(col + 32 + 8 * h);
  int2 b67 = *(const int2*)(col + 48 + 8 * h);
  v8i b;
  b[0] = b01.x; b[1] = b01.y; b[2] = b23.x; b[3] = b23.y;
  b[4] = b45.x; b[5] = b45.y; b[6] = b67.x; b[7] = b67.y;
  return b;
}

__device__ __forceinline__ unsigned pkbyte(int v, int sh) {
  return ((unsigned)(unsigned char)v) << sh;
}

__global__ __launch_bounds__(256) void pack_weights(
    const float* __restrict__ w1, const float* __restrict__ w2,
    const float* __restrict__ w3,
    const float* __restrict__ bn1w, const float* __restrict__ bn1b,
    const float* __restrict__ bn2w, const float* __restrict__ bn2b,
    const float* __restrict__ bn3w, const float* __restrict__ bn3b,
    const int* __restrict__ conv_zout, signed char* __restrict__ ws) {
  int i = blockIdx.x * blockDim.x + threadIdx.x;
  if (i < 64 * 256) ws[OFF_W1 + i] = (signed char)__float2int_rn(w1[i]);
  if (i < 64 * 576) {
    int cout = i / 576, rem = i % 576, tap = rem / 64, ci = rem % 64;
    ws[OFF_W2 + i] = (signed char)__float2int_rn(w2[(cout * 64 + ci) * 9 + tap]);
  }
  if (i < 256 * 64) ws[OFF_W3 + i] = (signed char)__float2int_rn(w3[i]);
  int* bni = (int*)(ws + OFF_BN);
  if (i < 64) {
    int wv = __float2int_rn(bn1w[i]);
    bni[BN1W + i] = wv;
    bni[BN1B + i] = __float2int_rn(bn1b[i]) - conv_zout[0] * wv;
    int wv2 = __float2int_rn(bn2w[i]);
    bni[BN2W + i] = wv2;
    bni[BN2B + i] = __float2int_rn(bn2b[i]) - conv_zout[1] * wv2;
  }
  if (i < 256) {
    int wv = __float2int_rn(bn3w[i]);
    bni[BN3W + i] = wv;
    bni[BN3B + i] = __float2int_rn(bn3b[i]) - conv_zout[2] * wv;
  }
}

__global__ __launch_bounds__(256) void conv1_bn1(
    const int* __restrict__ x, const signed char* __restrict__ A1,
    const int* __restrict__ bni, signed char* __restrict__ y1,
    signed char* __restrict__ ra,
    const int* conv_zin, const int* conv_m0, const int* conv_shift, const int* conv_zout,
    const int* bn_m0, const int* bn_shift, const int* bn_zout,
    const int* add_z, const int* add_m0, const int* add_shift) {
  __shared__ signed char Bt[64 * 256];
  __shared__ unsigned char raL[256];
  const int t   = threadIdx.x;
  const int q0  = blockIdx.x * 64;
  const int b   = q0 / 3136;
  const int hw0 = q0 % 3136;
  const int zin = conv_zin[0];
  raL[t] = (unsigned char)m0shift32(t - add_z[0], add_m0[0], add_shift[0]);
  __syncthreads();
  {
    int n = t & 63, g = t >> 6;
    const int* xp = x + (size_t)(b * 256) * 3136 + hw0 + n;
    signed char* rap = ra + (size_t)(b * 3136 + hw0 + n) * 256;
    for (int it = 0; it < 16; ++it) {
      int c = g * 4 + it * 16;
      int v0 = xp[(size_t)(c + 0) * 3136];
      int v1 = xp[(size_t)(c + 1) * 3136];
      int v2 = xp[(size_t)(c + 2) * 3136];
      int v3 = xp[(size_t)(c + 3) * 3136];
      unsigned pb = pkbyte(v0 - zin, 0) | pkbyte(v1 - zin, 8) |
                    pkbyte(v2 - zin, 16) | pkbyte(v3 - zin, 24);
      *(unsigned*)(Bt + n * 256 + c) = pb;
      unsigned pr = (unsigned)raL[v0] | ((unsigned)raL[v1] << 8) |
                    ((unsigned)raL[v2] << 16) | ((unsigned)raL[v3] << 24);
      *(unsigned*)(rap + c) = pr;
    }
  }
  __syncthreads();
  const int lane = t & 31, wv = t >> 5, col = lane & 15, h = lane >> 4;
  const int M0c = conv_m0[0], shc = conv_shift[0], zoc = conv_zout[0];
  const int M0b = bn_m0[0], shb = bn_shift[0], zob = bn_zout[0];
  const int zn = conv_zin[1];
  for (int tt = 0; tt < 2; ++tt) {
    int tile = wv * 2 + tt;
    int m0 = (tile >> 2) * 16, n0 = (tile & 3) * 16;
    v8i acc = {0, 0, 0, 0, 0, 0, 0, 0};
    for (int ks = 0; ks < 4; ++ks) {
      v8i a = load_a_frag(A1 + (m0 + col) * 256 + ks * 64, h);
      v8i bb = load_b_frag(Bt + (n0 + col) * 256 + ks * 64, h);
      acc = __builtin_amdgcn_wmma_i32_16x16x64_iu8(true, a, true, bb, acc, false, false);
    }
    const int* bw = bni + BN1W + m0 + 8 * h;
    const int* bbias = bni + BN1B + m0 + 8 * h;
    unsigned pk0 = 0, pk1 = 0;
#pragma unroll
    for (int r = 0; r < 8; ++r) {
      int q = rq32(acc[r], M0c, shc, zoc);
      int a2 = q * bw[r] + bbias[r];
      int qb = rq32(a2, M0b, shb, zob);
      if (r < 4) pk0 |= pkbyte(qb - zn, 8 * r);
      else       pk1 |= pkbyte(qb - zn, 8 * (r - 4));
    }
    int n = n0 + col;
    *(int2*)(y1 + (size_t)(b * 3136 + hw0 + n) * 64 + m0 + 8 * h) = make_int2(pk0, pk1);
  }
}

__global__ __launch_bounds__(256) void conv2_bn2(
    const signed char* __restrict__ y1, const signed char* __restrict__ A2,
    const int* __restrict__ bni, signed char* __restrict__ y2,
    const int* conv_zin, const int* conv_m0, const int* conv_shift, const int* conv_zout,
    const int* bn_m0, const int* bn_shift, const int* bn_zout) {
  __shared__ signed char Bt[64 * 576];
  const int t = threadIdx.x;
  int blk = blockIdx.x;
  int b = blk / 49, tyx = blk % 49;
  int h0 = (tyx / 7) * 8, w0 = (tyx % 7) * 8;
  for (int u = t; u < 576; u += 256) {
    int n = u / 9, tap = u % 9;
    int dh = tap / 3 - 1, dw = tap % 3 - 1;
    int hh = h0 + (n >> 3) + dh, ww = w0 + (n & 7) + dw;
    int4 d0, d1, d2, d3;
    if (hh >= 0 && hh < 56 && ww >= 0 && ww < 56) {
      const int4* src = (const int4*)(y1 + (size_t)(b * 3136 + hh * 56 + ww) * 64);
      d0 = src[0]; d1 = src[1]; d2 = src[2]; d3 = src[3];
    } else {
      d0 = d1 = d2 = d3 = make_int4(0, 0, 0, 0);
    }
    int4* dst = (int4*)(Bt + n * 576 + tap * 64);
    dst[0] = d0; dst[1] = d1; dst[2] = d2; dst[3] = d3;
  }
  __syncthreads();
  const int lane = t & 31, wv = t >> 5, col = lane & 15, h = lane >> 4;
  const int M0c = conv_m0[1], shc = conv_shift[1], zoc = conv_zout[1];
  const int M0b = bn_m0[1], shb = bn_shift[1], zob = bn_zout[1];
  const int zn = conv_zin[2];
  for (int tt = 0; tt < 2; ++tt) {
    int tile = wv * 2 + tt;
    int m0 = (tile >> 2) * 16, n0 = (tile & 3) * 16;
    v8i acc = {0, 0, 0, 0, 0, 0, 0, 0};
    for (int ks = 0; ks < 9; ++ks) {
      v8i a = load_a_frag(A2 + (m0 + col) * 576 + ks * 64, h);
      v8i bb = load_b_frag(Bt + (n0 + col) * 576 + ks * 64, h);
      acc = __builtin_amdgcn_wmma_i32_16x16x64_iu8(true, a, true, bb, acc, false, false);
    }
    const int* bw = bni + BN2W + m0 + 8 * h;
    const int* bbias = bni + BN2B + m0 + 8 * h;
    unsigned pk0 = 0, pk1 = 0;
#pragma unroll
    for (int r = 0; r < 8; ++r) {
      int q = rq32(acc[r], M0c, shc, zoc);
      int a2 = q * bw[r] + bbias[r];
      int qb = rq32(a2, M0b, shb, zob);
      if (r < 4) pk0 |= pkbyte(qb - zn, 8 * r);
      else       pk1 |= pkbyte(qb - zn, 8 * (r - 4));
    }
    int n = n0 + col;
    int hw = (h0 + (n >> 3)) * 56 + (w0 + (n & 7));
    *(int2*)(y2 + (size_t)(b * 3136 + hw) * 64 + m0 + 8 * h) = make_int2(pk0, pk1);
  }
}

__global__ __launch_bounds__(256) void conv3_bn3_add(
    const signed char* __restrict__ y2, const signed char* __restrict__ A3,
    const int* __restrict__ bni, const signed char* __restrict__ ra,
    int* __restrict__ out,
    const int* conv_m0, const int* conv_shift, const int* conv_zout,
    const int* bn_m0, const int* bn_shift, const int* bn_zout,
    const int* add_z, const int* add_m0, const int* add_shift, const int* add_zout) {
  __shared__ signed char Bt[64 * 64];
  __shared__ int rbl[256];
  const int t = threadIdx.x;
  const int q0 = blockIdx.x * 64;
  const int b = q0 / 3136, hw0 = q0 % 3136;
  rbl[t] = m0shift32(t - add_z[1], add_m0[1], add_shift[1]);
  {
    int n = t >> 2, chunk = t & 3;
    const int4* src = (const int4*)(y2 + (size_t)(b * 3136 + hw0 + n) * 64 + chunk * 16);
    *(int4*)(Bt + n * 64 + chunk * 16) = *src;
  }
  __syncthreads();
  const int lane = t & 31, wv = t >> 5, col = lane & 15, h = lane >> 4;
  const int M0c = conv_m0[2], shc = conv_shift[2], zoc = conv_zout[2];
  const int M0b = bn_m0[2], shb = bn_shift[2], zob = bn_zout[2];
  const int azo = add_zout[0];
  const signed char* rap = ra + (size_t)(b * 3136 + hw0) * 256;
  for (int tt = 0; tt < 8; ++tt) {
    int tile = wv * 8 + tt;
    int m0 = (tile >> 2) * 16, n0 = (tile & 3) * 16;
    v8i a = load_a_frag(A3 + (m0 + col) * 64, h);
    v8i bb = load_b_frag(Bt + (n0 + col) * 64, h);
    v8i acc = {0, 0, 0, 0, 0, 0, 0, 0};
    acc = __builtin_amdgcn_wmma_i32_16x16x64_iu8(true, a, true, bb, acc, false, false);
    const int* bw = bni + BN3W + m0 + 8 * h;
    const int* bbias = bni + BN3B + m0 + 8 * h;
    int n = n0 + col;
    int2 rpk = *(const int2*)(rap + (size_t)n * 256 + m0 + 8 * h);
#pragma unroll
    for (int r = 0; r < 8; ++r) {
      int m = m0 + r + 8 * h;
      int q = rq32(acc[r], M0c, shc, zoc);
      int a2 = q * bw[r] + bbias[r];
      int qb = rq32(a2, M0b, shb, zob);
      int rb = rbl[qb];
      unsigned pk = (r < 4) ? (unsigned)rpk.x : (unsigned)rpk.y;
      int rav = (int)(pk << (24 - 8 * (r & 3))) >> 24;
      int res = rav + rb + azo;
      res = res < 0 ? 0 : res;
      res = res > 255 ? 255 : res;
      out[(size_t)(b * 256 + m) * 3136 + hw0 + n] = res;
    }
  }
}

extern "C" void kernel_launch(void* const* d_in, const int* in_sizes, int n_in,
                              void* d_out, int out_size, void* d_ws, size_t ws_size,
                              hipStream_t stream) {
  const int*   x    = (const int*)d_in[0];
  const float* w1   = (const float*)d_in[1];
  const float* w2   = (const float*)d_in[2];
  const float* w3   = (const float*)d_in[3];
  const float* bn1w = (const float*)d_in[4];
  const float* bn1b = (const float*)d_in[5];
  const float* bn2w = (const float*)d_in[6];
  const float* bn2b = (const float*)d_in[7];
  const float* bn3w = (const float*)d_in[8];
  const float* bn3b = (const float*)d_in[9];
  const int* conv_zin   = (const int*)d_in[10];
  const int* conv_m0    = (const int*)d_in[11];
  const int* conv_shift = (const int*)d_in[12];
  const int* conv_zout  = (const int*)d_in[13];
  const int* bn_m0      = (const int*)d_in[14];
  const int* bn_shift   = (const int*)d_in[15];
  const int* bn_zout    = (const int*)d_in[16];
  const int* add_z      = (const int*)d_in[17];
  const int* add_m0     = (const int*)d_in[18];
  const int* add_shift  = (const int*)d_in[19];
  const int* add_zout   = (const int*)d_in[20];
  signed char* ws = (signed char*)d_ws;
  const int* bni = (const int*)(ws + OFF_BN);
  int* out = (int*)d_out;

  pack_weights<<<144, 256, 0, stream>>>(w1, w2, w3, bn1w, bn1b, bn2w, bn2b,
                                        bn3w, bn3b, conv_zout, ws);
  conv1_bn1<<<1568, 256, 0, stream>>>(x, ws + OFF_W1, bni, ws + OFF_Y1, ws + OFF_RA,
      conv_zin, conv_m0, conv_shift, conv_zout, bn_m0, bn_shift, bn_zout,
      add_z, add_m0, add_shift);
  conv2_bn2<<<1568, 256, 0, stream>>>(ws + OFF_Y1, ws + OFF_W2, bni, ws + OFF_Y2,
      conv_zin, conv_m0, conv_shift, conv_zout, bn_m0, bn_shift, bn_zout);
  conv3_bn3_add<<<1568, 256, 0, stream>>>(ws + OFF_Y2, ws + OFF_W3, bni, ws + OFF_RA,
      out, conv_m0, conv_shift, conv_zout, bn_m0, bn_shift, bn_zout,
      add_z, add_m0, add_shift, add_zout);
}
